// StabilityPredictorAP_43009802502314
// MI455X (gfx1250) — compile-verified, matrix operand order checked by reading, not run on hardware
//
#include <hip/hip_runtime.h>
#include <math.h>

// ---------------------------------------------------------------------------
// StabilityPredictor forward for MI455X (gfx1250).
// Heavy edge-wise MLPs (B*L*K = 98304 rows, H=128) run on the WMMA pipe:
// v_wmma_f32_16x16x32_f16 with f16 activations/weights, f32 accumulation.
// Everything else (topk, RBF features, LN+residual fusions, softmax pool,
// head) is plain wave32 VALU work -- <5% of total FLOPs.
// ---------------------------------------------------------------------------

#define BB 4
#define LL 512
#define KK 48
#define HH 128
#define NEDGE (BB * LL * KK)   // 98304
#define NNODE (BB * LL)        // 2048
#define EDGE_IN 416            // 16 pos + 25*16 rbf

typedef _Float16 f16;
typedef __attribute__((ext_vector_type(16))) _Float16 v16h;
typedef __attribute__((ext_vector_type(8)))  _Float16 v8h;
typedef __attribute__((ext_vector_type(8)))  float    v8f;

__device__ __forceinline__ float gelu_f(float x) {
  return 0.5f * x * (1.0f + erff(x * 0.70710678118654752f));
}

// atom index map: N=0, Ca=1, C=2, O=3, Cb=4 ; 24 PAIRS from reference
__constant__ int c_pa[24] = {0,2,3,4,1,1,1,1,0,0,0,4,4,3,0,2,3,4,2,3,4,2,3,2};
__constant__ int c_pb[24] = {0,2,3,4,0,2,3,4,2,3,4,2,3,2,1,1,1,1,0,0,0,4,4,3};

// ---------------------------------------------------------------------------
// WMMA GEMM:  C[M,N] = A[M,Kd] (f16 row-major) * Bt[N,Kd]^T (weights stored
// N-major) + bias, optional exact GELU, output f16 or f32.
// One wave computes a 16(M) x 64(N) strip: A fragment reused across 4 WMMAs.
// Lane layout per ISA: lane = (half<<4)|r ; A frag elems 0..7 = K[k0+half*8..],
// elems 8..15 = K[k0+16+half*8..]; B frag identical K-order for column r.
// ---------------------------------------------------------------------------
template<bool GELU, bool HASB, bool OUT16>
__global__ void gemm_k(const f16* __restrict__ A, const f16* __restrict__ Bt,
                       const float* __restrict__ bias, void* __restrict__ Cv,
                       int M, int Kd, int N)
{
  const int wave = threadIdx.x >> 5;
  const int lane = threadIdx.x & 31;
  const int r  = lane & 15;
  const int hf = lane >> 4;
  const int tilesM = M >> 4;
  const int nG = N >> 6;                 // groups of 64 columns
  int job = blockIdx.x * (blockDim.x >> 5) + wave;
  if (job >= tilesM * nG) return;
  const int tm = job / nG;
  const int g  = job % nG;

  const f16* Arow = A + (size_t)(tm * 16 + r) * Kd;
  v8f acc[4] = {};
  for (int k0 = 0; k0 < Kd; k0 += 32) {
    v16h a;
    ((v8h*)&a)[0] = *(const v8h*)(Arow + k0 + hf * 8);
    ((v8h*)&a)[1] = *(const v8h*)(Arow + k0 + 16 + hf * 8);
#pragma unroll
    for (int t = 0; t < 4; ++t) {
      const f16* Brow = Bt + (size_t)(g * 64 + t * 16 + r) * Kd;
      v16h bf;
      ((v8h*)&bf)[0] = *(const v8h*)(Brow + k0 + hf * 8);
      ((v8h*)&bf)[1] = *(const v8h*)(Brow + k0 + 16 + hf * 8);
      acc[t] = __builtin_amdgcn_wmma_f32_16x16x32_f16(
          false, a, false, bf, (short)0, acc[t], false, false);
    }
  }
#pragma unroll
  for (int t = 0; t < 4; ++t) {
    const int col = g * 64 + t * 16 + r;
    const float bv = HASB ? bias[col] : 0.0f;
#pragma unroll
    for (int v = 0; v < 8; ++v) {
      const int row = tm * 16 + v + 8 * hf;   // C/D layout: VGPR v -> M=v+8*half
      float val = acc[t][v] + bv;
      if (GELU) val = gelu_f(val);
      if (OUT16) ((f16*)Cv)[(size_t)row * N + col] = (f16)val;
      else       ((float*)Cv)[(size_t)row * N + col] = val;
    }
  }
}

// ---------------------------------------------------------------------------
// small utility kernels
// ---------------------------------------------------------------------------
__global__ void wcvt_k(const float* __restrict__ W, f16* __restrict__ Wt,
                       int Kd, int N) {            // W[Kd][N] -> Wt[N][Kd] f16
  int i = blockIdx.x * blockDim.x + threadIdx.x;
  if (i >= Kd * N) return;
  int n = i / Kd, k = i % Kd;
  Wt[i] = (f16)W[(size_t)k * N + n];
}

__global__ void cvt_k(const float* __restrict__ s, f16* __restrict__ d, size_t n) {
  size_t i = (size_t)blockIdx.x * blockDim.x + threadIdx.x;
  if (i < n) d[i] = (f16)s[i];
}

__global__ void cb_k(const float* __restrict__ X, float* __restrict__ Cb) {
  int r = blockIdx.x * blockDim.x + threadIdx.x;
  if (r >= NNODE) return;
  const float* p = X + (size_t)r * 12;
  float bv[3], cv[3], av[3];
#pragma unroll
  for (int d = 0; d < 3; ++d) { bv[d] = p[3 + d] - p[d]; cv[d] = p[6 + d] - p[3 + d]; }
  av[0] = bv[1] * cv[2] - bv[2] * cv[1];
  av[1] = bv[2] * cv[0] - bv[0] * cv[2];
  av[2] = bv[0] * cv[1] - bv[1] * cv[0];
#pragma unroll
  for (int d = 0; d < 3; ++d)
    Cb[r * 3 + d] = -0.58273431f * av[d] + 0.56802827f * bv[d] - 0.54067466f * cv[d] + p[3 + d];
}

// one block (512 threads) per (b,i): masked distances + iterative top-K min
__global__ void topk_k(const float* __restrict__ X, const float* __restrict__ mask,
                       int* __restrict__ Eidx, float* __restrict__ Dn) {
  int b = blockIdx.x / LL, i = blockIdx.x % LL;
  int j = threadIdx.x;
  __shared__ float sv[LL];
  __shared__ float rv[LL];
  __shared__ int   ri[LL];
  const float* Ci = X + (((size_t)b * LL + i) * 4 + 1) * 3;
  const float* Cj = X + (((size_t)b * LL + j) * 4 + 1) * 3;
  float dx = Ci[0] - Cj[0], dy = Ci[1] - Cj[1], dz = Ci[2] - Cj[2];
  float m2 = mask[b * LL + i] * mask[b * LL + j];
  float D = m2 * sqrtf(dx * dx + dy * dy + dz * dz + 1e-6f);
  rv[j] = D; __syncthreads();
  for (int s = LL / 2; s > 0; s >>= 1) { if (j < s) rv[j] = fmaxf(rv[j], rv[j + s]); __syncthreads(); }
  float Dmax = rv[0]; __syncthreads();
  sv[j] = D + (1.0f - m2) * Dmax;
  __syncthreads();
  for (int k = 0; k < KK; ++k) {
    rv[j] = sv[j]; ri[j] = j; __syncthreads();
    for (int s = LL / 2; s > 0; s >>= 1) {
      if (j < s) {
        if (rv[j + s] < rv[j] || (rv[j + s] == rv[j] && ri[j + s] < ri[j])) {
          rv[j] = rv[j + s]; ri[j] = ri[j + s];
        }
      }
      __syncthreads();
    }
    if (j == 0) { Eidx[(size_t)blockIdx.x * KK + k] = ri[0]; Dn[(size_t)blockIdx.x * KK + k] = rv[0]; }
    __syncthreads();
    if (j == ri[0]) sv[j] = 3.4e38f;
    __syncthreads();
  }
}

// per edge: 16 pos features + 25 x 16 RBFs -> f16 row of width 416
__global__ void edgefeat_k(const float* __restrict__ X, const float* __restrict__ Cbb,
                           const int* __restrict__ Eidx, const float* __restrict__ Dn,
                           const int* __restrict__ resid, const int* __restrict__ chenc,
                           const float* __restrict__ posW, const float* __restrict__ posB,
                           f16* __restrict__ out) {
  int e = blockIdx.x * blockDim.x + threadIdx.x;
  if (e >= NEDGE) return;
  int bl = e / KK, b = bl / LL;
  int j = Eidx[e], blj = b * LL + j;
  float cc[5][3], nc[5][3];
#pragma unroll
  for (int a = 0; a < 4; ++a)
#pragma unroll
    for (int d = 0; d < 3; ++d) {
      cc[a][d] = X[((size_t)bl * 4 + a) * 3 + d];
      nc[a][d] = X[((size_t)blj * 4 + a) * 3 + d];
    }
#pragma unroll
  for (int d = 0; d < 3; ++d) { cc[4][d] = Cbb[bl * 3 + d]; nc[4][d] = Cbb[blj * 3 + d]; }

  f16* o = out + (size_t)e * EDGE_IN;
  int off = resid[bl] - resid[blj];
  int same = (chenc[bl] == chenc[blj]) ? 1 : 0;
  int dp = off + 32; dp = dp < 0 ? 0 : (dp > 64 ? 64 : dp);
  int dpos = same ? dp : 65;
#pragma unroll
  for (int t = 0; t < 16; ++t) o[t] = (f16)(posW[dpos * 16 + t] + posB[t]);
  float dist = Dn[e];
#pragma unroll
  for (int t = 0; t < 16; ++t) {
    float z = (dist - (2.0f + t * (20.0f / 15.0f))) * 0.8f;   // sigma = 1.25
    o[16 + t] = (f16)expf(-z * z);
  }
  for (int p = 0; p < 24; ++p) {
    int ia = c_pa[p], ib = c_pb[p];
    float dx = cc[ia][0] - nc[ib][0], dy = cc[ia][1] - nc[ib][1], dz = cc[ia][2] - nc[ib][2];
    float dd = sqrtf(dx * dx + dy * dy + dz * dz + 1e-6f);
#pragma unroll
    for (int t = 0; t < 16; ++t) {
      float z = (dd - (2.0f + t * (20.0f / 15.0f))) * 0.8f;
      o[32 + p * 16 + t] = (f16)expf(-z * z);
    }
  }
}

// block-of-128 LayerNorm helpers (H = 128)
__device__ __forceinline__ float blk_sum(float v, float* red, int t) {
  red[t] = v; __syncthreads();
#pragma unroll
  for (int s = 64; s > 0; s >>= 1) { if (t < s) red[t] += red[t + s]; __syncthreads(); }
  float r = red[0]; __syncthreads();
  return r;
}
__device__ __forceinline__ float ln_val(float v, float g, float b, float* red, int t) {
  float mean = blk_sum(v, red, t) * (1.0f / HH);
  float d = v - mean;
  float var = blk_sum(d * d, red, t) * (1.0f / HH);
  return d * rsqrtf(var + 1e-5f) * g + b;
}

__global__ void ln_rows_k(const float* __restrict__ x, const float* __restrict__ g,
                          const float* __restrict__ b, float* __restrict__ y32,
                          f16* __restrict__ y16) {
  size_t r = blockIdx.x; int t = threadIdx.x;
  __shared__ float red[HH];
  float y = ln_val(x[r * HH + t], g[t], b[t], red, t);
  if (y32) y32[r * HH + t] = y;
  if (y16) y16[r * HH + t] = (f16)y;
}

// h_V = LN(h_V + sum_k(w_k * msg)/30); optional mask_attend weight
__global__ void node_upd1_k(const float* __restrict__ msg, const int* __restrict__ Eidx,
                            const float* __restrict__ mask, const float* __restrict__ hVin,
                            const float* __restrict__ g, const float* __restrict__ b,
                            float* __restrict__ hV32, f16* __restrict__ hV16, int useAtt) {
  int r = blockIdx.x; int t = threadIdx.x; int bb = r / LL;
  __shared__ float red[HH];
  float s = 0.0f;
  for (int k = 0; k < KK; ++k) {
    size_t e = (size_t)r * KK + k;
    float w = 1.0f;
    if (useAtt) w = mask[r] * mask[bb * LL + Eidx[e]];
    s += w * msg[e * HH + t];
  }
  float v = hVin[(size_t)r * HH + t] + s * (1.0f / 30.0f);
  float y = ln_val(v, g[t], b[t], red, t);
  hV32[(size_t)r * HH + t] = y;
  hV16[(size_t)r * HH + t] = (f16)y;
}

// h_V = mask * LN(h_V + ffn)
__global__ void node_upd2_k(const float* __restrict__ ffn, const float* __restrict__ hVin,
                            const float* __restrict__ mask, const float* __restrict__ g,
                            const float* __restrict__ b, float* __restrict__ hV32,
                            f16* __restrict__ hV16) {
  int r = blockIdx.x; int t = threadIdx.x;
  __shared__ float red[HH];
  float v = hVin[(size_t)r * HH + t] + ffn[(size_t)r * HH + t];
  float y = ln_val(v, g[t], b[t], red, t) * mask[r];
  hV32[(size_t)r * HH + t] = y;
  hV16[(size_t)r * HH + t] = (f16)y;
}

// h_E = LN(h_E + msgE) (in-place f32 + f16 copy)
__global__ void edge_upd_k(const float* __restrict__ msgE, float* __restrict__ hE32,
                           const float* __restrict__ g, const float* __restrict__ b,
                           f16* __restrict__ hE16) {
  size_t r = blockIdx.x; int t = threadIdx.x;
  __shared__ float red[HH];
  float v = hE32[r * HH + t] + msgE[r * HH + t];
  float y = ln_val(v, g[t], b[t], red, t);
  hE32[r * HH + t] = y;
  hE16[r * HH + t] = (f16)y;
}

// encoder concat input: [h_V(center) | h_E | h_V(neighbor)]  (384 wide)
__global__ void build_hEV_k(const f16* __restrict__ hV16, const f16* __restrict__ hE16,
                            const int* __restrict__ Eidx, f16* __restrict__ out) {
  size_t e = blockIdx.x; int t = threadIdx.x;
  int bl = (int)(e / KK), b = bl / LL;
  int j = Eidx[e];
  f16* o = out + e * (3 * HH);
  o[t]            = hV16[(size_t)bl * HH + t];
  o[HH + t]       = hE16[e * HH + t];
  o[2 * HH + t]   = hV16[((size_t)b * LL + j) * HH + t];
}

// decoder concat input (512 wide):
// [hV_cur | m1*hE | m1*att*hS_n | m1*(att*hV_cur_n + (1-att)*hV_enc_n)]
__global__ void build_dec_k(const f16* __restrict__ hV16, const f16* __restrict__ hE16,
                            const f16* __restrict__ hS16, const f16* __restrict__ hVe16,
                            const int* __restrict__ Eidx, const int* __restrict__ ranks,
                            const float* __restrict__ mask, f16* __restrict__ out) {
  size_t e = blockIdx.x; int t = threadIdx.x;
  int bl = (int)(e / KK), b = bl / LL;
  int j = Eidx[e]; size_t nj = (size_t)b * LL + j;
  float att = (ranks[bl] > ranks[b * LL + j]) ? 1.0f : 0.0f;
  float m1 = mask[bl];
  f16* o = out + e * (4 * HH);
  o[t]            = hV16[(size_t)bl * HH + t];
  o[HH + t]       = (f16)(m1 * (float)hE16[e * HH + t]);
  o[2 * HH + t]   = (f16)(m1 * att * (float)hS16[nj * HH + t]);
  o[3 * HH + t]   = (f16)(m1 * (att * (float)hV16[nj * HH + t] +
                                 (1.0f - att) * (float)hVe16[nj * HH + t]));
}

// ranks[i] = #(key[j] < key[i]) + #(key[j]==key[i] && j<i)  (== argsort(argsort))
__global__ void ranks_k(const float* __restrict__ chainM, const float* __restrict__ noise,
                        int* __restrict__ ranks) {
  int b = blockIdx.x, i = threadIdx.x;
  __shared__ float key[LL];
  float ki = (chainM[b * LL + i] + 1e-4f) * fabsf(noise[b * LL + i]);
  key[i] = ki; __syncthreads();
  int c = 0;
  for (int j = 0; j < LL; ++j) {
    float kj = key[j];
    c += (kj < ki) || (kj == ki && j < i);
  }
  ranks[b * LL + i] = c;
}

__global__ void gather_embed_k(const int* __restrict__ S, const float* __restrict__ Ws,
                               f16* __restrict__ hS16) {
  int i = blockIdx.x * blockDim.x + threadIdx.x;
  if (i >= NNODE * HH) return;
  int r = i / HH, h = i % HH;
  hS16[i] = (f16)Ws[(size_t)S[r] * HH + h];
}

// attention scores: a = tanh(hV@W1+b1)@W2 + b2 ; masked -> -1e9
__global__ void attn_k(const float* __restrict__ hV, const float* __restrict__ W1,
                       const float* __restrict__ b1, const float* __restrict__ W2,
                       const float* __restrict__ b2, const float* __restrict__ mask,
                       float* __restrict__ a) {
  int r = blockIdx.x; int u = threadIdx.x;   // 64 threads
  __shared__ float red[64];
  float s = 0.0f;
  for (int h = 0; h < HH; ++h) s += hV[(size_t)r * HH + h] * W1[h * 64 + u];
  float hid = tanhf(s + b1[u]);
  red[u] = hid * W2[u]; __syncthreads();
#pragma unroll
  for (int st = 32; st > 0; st >>= 1) { if (u < st) red[u] += red[u + st]; __syncthreads(); }
  if (u == 0) a[r] = (mask[r] == 0.0f) ? -1e9f : (red[0] + b2[0]);
}

// softmax over L + weighted pool of hV
__global__ void pool_k(const float* __restrict__ a, const float* __restrict__ hV,
                       float* __restrict__ gout) {
  int b = blockIdx.x; int t = threadIdx.x;   // 128 threads
  __shared__ float red[HH];
  __shared__ float w[LL];
  float m = -3.4e38f;
  for (int l = t; l < LL; l += HH) m = fmaxf(m, a[b * LL + l]);
  red[t] = m; __syncthreads();
#pragma unroll
  for (int s = 64; s > 0; s >>= 1) { if (t < s) red[t] = fmaxf(red[t], red[t + s]); __syncthreads(); }
  float amax = red[0]; __syncthreads();
  float ssum = 0.0f;
  for (int l = t; l < LL; l += HH) { float e1 = expf(a[b * LL + l] - amax); w[l] = e1; ssum += e1; }
  float tot = blk_sum(ssum, red, t);
  float acc = 0.0f;
  for (int l = 0; l < LL; ++l) acc += w[l] * hV[((size_t)b * LL + l) * HH + t];
  gout[b * HH + t] = acc / tot;
}

// head: gelu(g@W1+b1) -> gelu(@W2+b2) -> @W3+b3 -> out[b]
__global__ void head_k(const float* __restrict__ gin, const float* __restrict__ W1,
                       const float* __restrict__ b1, const float* __restrict__ W2,
                       const float* __restrict__ b2, const float* __restrict__ W3,
                       const float* __restrict__ b3, float* __restrict__ out) {
  int b = blockIdx.x; int t = threadIdx.x;   // 128 threads
  __shared__ float x1[HH];
  __shared__ float x2[64];
  __shared__ float red[64];
  float s = 0.0f;
  for (int h = 0; h < HH; ++h) s += gin[b * HH + h] * W1[h * HH + t];
  x1[t] = gelu_f(s + b1[t]); __syncthreads();
  if (t < 64) {
    float s2 = 0.0f;
    for (int h = 0; h < HH; ++h) s2 += x1[h] * W2[h * 64 + t];
    x2[t] = gelu_f(s2 + b2[t]);
  }
  __syncthreads();
  if (t < 64) red[t] = x2[t] * W3[t];
  __syncthreads();
  if (t < 64) {
#pragma unroll
    for (int st = 32; st > 0; st >>= 1) { if (t < st) red[t] += red[t + st]; __syncthreads(); }
  } else {
#pragma unroll
    for (int st = 32; st > 0; st >>= 1) { __syncthreads(); }
  }
  if (t == 0) out[b] = red[0] + b3[0];
}

// ---------------------------------------------------------------------------
static void launch_gemm(const f16* A, const f16* Bt, const float* bias, void* C,
                        int M, int Kd, int N, bool gelu, bool out16, hipStream_t s) {
  int jobs = (M >> 4) * (N >> 6);
  dim3 grd((jobs + 3) / 4), blk(128);
  if (gelu && bias && out16)       gemm_k<true , true , true ><<<grd, blk, 0, s>>>(A, Bt, bias, C, M, Kd, N);
  else if (gelu && bias)           gemm_k<true , true , false><<<grd, blk, 0, s>>>(A, Bt, bias, C, M, Kd, N);
  else if (!gelu && bias && out16) gemm_k<false, true , true ><<<grd, blk, 0, s>>>(A, Bt, bias, C, M, Kd, N);
  else if (!gelu && bias)          gemm_k<false, true , false><<<grd, blk, 0, s>>>(A, Bt, bias, C, M, Kd, N);
  else if (!gelu && out16)         gemm_k<false, false, true ><<<grd, blk, 0, s>>>(A, Bt, bias, C, M, Kd, N);
  else                             gemm_k<false, false, false><<<grd, blk, 0, s>>>(A, Bt, bias, C, M, Kd, N);
}

// Input index map: recursive insertion-order flatten of setup_inputs():
// 0:X 1:S 2:mask 3:chain_M 4:residue_idx 5:chain_enc 6:noise,
// 7:pos_W 8:pos_b 9:edge_emb_W 10:norm_edges_g 11:norm_edges_b 12:We_W 13:We_b 14:W_s,
// enc layers (22 each, from 15): nW1,nb1,nW2,nb2,nW3,nb3,n1g,n1b,n2g,n2b,
//   fWi,fbi,fWo,fbo,eW1,eb1,eW2,eb2,eW3,eb3,n3g,n3b
// dec layers (14 each, from 81): nW1,nb1,nW2,nb2,nW3,nb3,n1g,n1b,n2g,n2b,fWi,fbi,fWo,fbo
// 123:attn_W1 124:attn_b1 125:attn_W2 126:attn_b2 127..132: head_W1..head_b3
extern "C" void kernel_launch(void* const* d_in, const int* in_sizes, int n_in,
                              void* d_out, int out_size, void* d_ws, size_t ws_size,
                              hipStream_t stream) {
  (void)in_sizes; (void)n_in; (void)out_size; (void)ws_size;
  const float* X      = (const float*)d_in[0];
  const int*   S      = (const int*)  d_in[1];
  const float* mask   = (const float*)d_in[2];
  const float* chainM = (const float*)d_in[3];
  const int*   resid  = (const int*)  d_in[4];
  const int*   chenc  = (const int*)  d_in[5];
  const float* noise  = (const float*)d_in[6];
  const float* posW   = (const float*)d_in[7];
  const float* posB   = (const float*)d_in[8];
  const float* normEg = (const float*)d_in[10];
  const float* normEb = (const float*)d_in[11];
  const float* WeB    = (const float*)d_in[13];
  const float* Ws     = (const float*)d_in[14];
  const int IENC = 15, ES = 22;
  const int IDEC = IENC + 3 * ES;        // 81
  const int IA   = IDEC + 3 * 14;        // 123

  // ---- workspace layout ----
  char* ws = (char*)d_ws;
  size_t off = 0;
  auto take = [&](size_t bytes) -> void* {
    void* p = ws + off; off += (bytes + 255) & ~(size_t)255; return p;
  };
  f16*   bufA  = (f16*)  take((size_t)NEDGE * 512 * sizeof(f16));   // concat GEMM inputs
  float* msg32 = (float*)take((size_t)NEDGE * HH * sizeof(float));  // f32 GEMM outputs
  float* hE32  = (float*)take((size_t)NEDGE * HH * sizeof(float));
  f16*   hE16  = (f16*)  take((size_t)NEDGE * HH * sizeof(f16));
  f16*   T1    = (f16*)  take((size_t)NEDGE * HH * sizeof(f16));
  f16*   T2    = (f16*)  take((size_t)NEDGE * HH * sizeof(f16));
  float* hV32  = (float*)take((size_t)NNODE * HH * sizeof(float));
  f16*   hV16  = (f16*)  take((size_t)NNODE * HH * sizeof(f16));
  f16*   hVe16 = (f16*)  take((size_t)NNODE * HH * sizeof(f16));
  f16*   hS16  = (f16*)  take((size_t)NNODE * HH * sizeof(f16));
  float* Cbb   = (float*)take((size_t)NNODE * 3 * sizeof(float));
  int*   Eidx  = (int*)  take((size_t)NEDGE * sizeof(int));
  float* Dn    = (float*)take((size_t)NEDGE * sizeof(float));
  int*   ranks = (int*)  take((size_t)NNODE * sizeof(int));
  float* ascore= (float*)take((size_t)NNODE * sizeof(float));
  float* gpool = (float*)take((size_t)BB * HH * sizeof(float));
  f16*   wbase = (f16*)  take((size_t)4 * 1024 * 1024);             // f16 transposed weights

  // ---- weights: f32 -> transposed f16 ----
  f16* wp = wbase;
  auto cvtW = [&](int idx, int Kd, int N) -> const f16* {
    f16* dst = wp; wp += (size_t)Kd * N;
    int n = Kd * N;
    wcvt_k<<<dim3((n + 255) / 256), dim3(256), 0, stream>>>((const float*)d_in[idx], dst, Kd, N);
    return dst;
  };
  const f16* edgeWT = cvtW(9,  EDGE_IN, HH);
  const f16* WeWT   = cvtW(12, HH, HH);
  struct EncW { const f16 *nW1,*nW2,*nW3,*fWi,*fWo,*eW1,*eW2,*eW3; } encw[3];
  for (int l = 0; l < 3; ++l) {
    int b = IENC + l * ES;
    encw[l].nW1 = cvtW(b + 0,  3 * HH, HH);
    encw[l].nW2 = cvtW(b + 2,  HH, HH);
    encw[l].nW3 = cvtW(b + 4,  HH, HH);
    encw[l].fWi = cvtW(b + 10, HH, 4 * HH);
    encw[l].fWo = cvtW(b + 12, 4 * HH, HH);
    encw[l].eW1 = cvtW(b + 14, 3 * HH, HH);
    encw[l].eW2 = cvtW(b + 16, HH, HH);
    encw[l].eW3 = cvtW(b + 18, HH, HH);
  }
  struct DecW { const f16 *nW1,*nW2,*nW3,*fWi,*fWo; } decw[3];
  for (int l = 0; l < 3; ++l) {
    int b = IDEC + l * 14;
    decw[l].nW1 = cvtW(b + 0,  4 * HH, HH);
    decw[l].nW2 = cvtW(b + 2,  HH, HH);
    decw[l].nW3 = cvtW(b + 4,  HH, HH);
    decw[l].fWi = cvtW(b + 10, HH, 4 * HH);
    decw[l].fWo = cvtW(b + 12, 4 * HH, HH);
  }

  // ---- protein features ----
  cb_k<<<dim3((NNODE + 127) / 128), dim3(128), 0, stream>>>(X, Cbb);
  topk_k<<<dim3(NNODE), dim3(LL), 0, stream>>>(X, mask, Eidx, Dn);
  edgefeat_k<<<dim3((NEDGE + 127) / 128), dim3(128), 0, stream>>>(
      X, Cbb, Eidx, Dn, resid, chenc, posW, posB, bufA);
  launch_gemm(bufA, edgeWT, nullptr, msg32, NEDGE, EDGE_IN, HH, false, false, stream);
  ln_rows_k<<<dim3(NEDGE), dim3(HH), 0, stream>>>(msg32, normEg, normEb, nullptr, T1);
  launch_gemm(T1, WeWT, WeB, hE32, NEDGE, HH, HH, false, false, stream);
  cvt_k<<<dim3((unsigned)(((size_t)NEDGE * HH + 255) / 256)), dim3(256), 0, stream>>>(
      hE32, hE16, (size_t)NEDGE * HH);
  hipMemsetAsync(hV32, 0, (size_t)NNODE * HH * sizeof(float), stream);
  hipMemsetAsync(hV16, 0, (size_t)NNODE * HH * sizeof(f16), stream);

  // ---- encoder layers ----
  for (int l = 0; l < 3; ++l) {
    int b = IENC + l * ES;
    const float* nb1 = (const float*)d_in[b + 1];
    const float* nb2 = (const float*)d_in[b + 3];
    const float* nb3 = (const float*)d_in[b + 5];
    const float* n1g = (const float*)d_in[b + 6];
    const float* n1b = (const float*)d_in[b + 7];
    const float* n2g = (const float*)d_in[b + 8];
    const float* n2b = (const float*)d_in[b + 9];
    const float* fbi = (const float*)d_in[b + 11];
    const float* fbo = (const float*)d_in[b + 13];
    const float* eb1 = (const float*)d_in[b + 15];
    const float* eb2 = (const float*)d_in[b + 17];
    const float* eb3 = (const float*)d_in[b + 19];
    const float* n3g = (const float*)d_in[b + 20];
    const float* n3b = (const float*)d_in[b + 21];

    build_hEV_k<<<dim3(NEDGE), dim3(HH), 0, stream>>>(hV16, hE16, Eidx, bufA);
    launch_gemm(bufA, encw[l].nW1, nb1, T1, NEDGE, 3 * HH, HH, true, true, stream);
    launch_gemm(T1, encw[l].nW2, nb2, T2, NEDGE, HH, HH, true, true, stream);
    launch_gemm(T2, encw[l].nW3, nb3, msg32, NEDGE, HH, HH, false, false, stream);
    node_upd1_k<<<dim3(NNODE), dim3(HH), 0, stream>>>(msg32, Eidx, mask, hV32, n1g, n1b, hV32, hV16, 1);
    launch_gemm(hV16, encw[l].fWi, fbi, T1, NNODE, HH, 4 * HH, true, true, stream);
    launch_gemm(T1, encw[l].fWo, fbo, msg32, NNODE, 4 * HH, HH, false, false, stream);
    node_upd2_k<<<dim3(NNODE), dim3(HH), 0, stream>>>(msg32, hV32, mask, n2g, n2b, hV32, hV16);
    build_hEV_k<<<dim3(NEDGE), dim3(HH), 0, stream>>>(hV16, hE16, Eidx, bufA);
    launch_gemm(bufA, encw[l].eW1, eb1, T1, NEDGE, 3 * HH, HH, true, true, stream);
    launch_gemm(T1, encw[l].eW2, eb2, T2, NEDGE, HH, HH, true, true, stream);
    launch_gemm(T2, encw[l].eW3, eb3, msg32, NEDGE, HH, HH, false, false, stream);
    edge_upd_k<<<dim3(NEDGE), dim3(HH), 0, stream>>>(msg32, hE32, n3g, n3b, hE16);
  }

  // ---- decoder prep ----
  ranks_k<<<dim3(BB), dim3(LL), 0, stream>>>(chainM, noise, ranks);
  gather_embed_k<<<dim3((NNODE * HH + 255) / 256), dim3(256), 0, stream>>>(S, Ws, hS16);
  hipMemcpyAsync(hVe16, hV16, (size_t)NNODE * HH * sizeof(f16),
                 hipMemcpyDeviceToDevice, stream);

  // ---- decoder layers ----
  for (int l = 0; l < 3; ++l) {
    int b = IDEC + l * 14;
    const float* nb1 = (const float*)d_in[b + 1];
    const float* nb2 = (const float*)d_in[b + 3];
    const float* nb3 = (const float*)d_in[b + 5];
    const float* n1g = (const float*)d_in[b + 6];
    const float* n1b = (const float*)d_in[b + 7];
    const float* n2g = (const float*)d_in[b + 8];
    const float* n2b = (const float*)d_in[b + 9];
    const float* fbi = (const float*)d_in[b + 11];
    const float* fbo = (const float*)d_in[b + 13];

    build_dec_k<<<dim3(NEDGE), dim3(HH), 0, stream>>>(hV16, hE16, hS16, hVe16, Eidx, ranks, mask, bufA);
    launch_gemm(bufA, decw[l].nW1, nb1, T1, NEDGE, 4 * HH, HH, true, true, stream);
    launch_gemm(T1, decw[l].nW2, nb2, T2, NEDGE, HH, HH, true, true, stream);
    launch_gemm(T2, decw[l].nW3, nb3, msg32, NEDGE, HH, HH, false, false, stream);
    node_upd1_k<<<dim3(NNODE), dim3(HH), 0, stream>>>(msg32, Eidx, mask, hV32, n1g, n1b, hV32, hV16, 0);
    launch_gemm(hV16, decw[l].fWi, fbi, T1, NNODE, HH, 4 * HH, true, true, stream);
    launch_gemm(T1, decw[l].fWo, fbo, msg32, NNODE, 4 * HH, HH, false, false, stream);
    node_upd2_k<<<dim3(NNODE), dim3(HH), 0, stream>>>(msg32, hV32, mask, n2g, n2b, hV32, hV16);
  }

  // ---- attention pooling + head ----
  attn_k<<<dim3(NNODE), dim3(64), 0, stream>>>(
      hV32, (const float*)d_in[IA + 0], (const float*)d_in[IA + 1],
      (const float*)d_in[IA + 2], (const float*)d_in[IA + 3], mask, ascore);
  pool_k<<<dim3(BB), dim3(HH), 0, stream>>>(ascore, hV32, gpool);
  head_k<<<dim3(BB), dim3(HH), 0, stream>>>(
      gpool, (const float*)d_in[IA + 4], (const float*)d_in[IA + 5],
      (const float*)d_in[IA + 6], (const float*)d_in[IA + 7],
      (const float*)d_in[IA + 8], (const float*)d_in[IA + 9], (float*)d_out);
}
